// velearner_82669530513592
// MI455X (gfx1250) — matrix operand order repaired and checked by reading, not compiled, not run on hardware
//
#include <hip/hip_runtime.h>
#include <math.h>

typedef __attribute__((ext_vector_type(16))) _Float16 v16h;
typedef __attribute__((ext_vector_type(8)))  _Float16 v8h;
typedef __attribute__((ext_vector_type(8)))  float    v8f;

#define NLVL  16
#define TBITS 19
#define TSIZE (1u << TBITS)

struct EncParams { int res[NLVL]; int dense[NLVL]; };

union H16 { v16h v; v8h h[2]; };

// One block = 256 threads = 8 waves; each wave owns a 16-point tile (128 pts/block).
__global__ __launch_bounds__(256, 1) void ngp_fused_kernel(
    const float* __restrict__ x,
    const float* __restrict__ tables,
    const float* __restrict__ v1, const float* __restrict__ g1, const float* __restrict__ b1,
    const float* __restrict__ v2, const float* __restrict__ g2, const float* __restrict__ b2,
    float* __restrict__ out, int npts, EncParams P)
{
  __shared__ __attribute__((aligned(16))) float    xsn[128][3];     // normalized coords
  __shared__ __attribute__((aligned(16))) _Float16 w1h[64][32];     // W1 (weight-normed), row = out unit
  __shared__ __attribute__((aligned(16))) _Float16 w2h[16][64];     // W2 padded to 16 rows
  __shared__ float b1s[64];
  __shared__ float b2s[4];
  __shared__ __attribute__((aligned(16))) _Float16 htile[8][16][32]; // per-wave encoded features
  __shared__ __attribute__((aligned(16))) _Float16 r1t[8][16][64];   // per-wave relu(layer1)

  const int t    = threadIdx.x;
  const int wave = t >> 5;
  const int lane = t & 31;
  const int hi   = (lane >> 4) & 1;   // 0: lanes 0-15, 1: lanes 16-31
  const int nIdx = lane & 15;
  const int blockBase = blockIdx.x * 128;

  // ---------- Phase 1: weight-norm prep + point load ----------
  {
    _Float16* wz = &w2h[0][0];        // zero the padded W2 (16*64 = 1024 halfs)
    #pragma unroll
    for (int q = 0; q < 4; ++q) wz[t * 4 + q] = (_Float16)0.0f;
  }
  if (t < 64) {
    float s = 0.0f;
    for (int i = 0; i < 32; ++i) { float v = v1[t * 32 + i]; s += v * v; }
    float sc = g1[t] / sqrtf(s);
    for (int i = 0; i < 32; ++i) w1h[t][i] = (_Float16)(v1[t * 32 + i] * sc);
    b1s[t] = b1[t];
  }
  if (t < 4) b2s[t] = (t < 3) ? b2[t] : 0.0f;
  if (t < 128) {
    int p = blockBase + t;
    if (p < npts) {
      xsn[t][0] = (x[p * 3 + 0] + 1.0f) * 0.5f;
      xsn[t][1] = (x[p * 3 + 1] + 1.0f) * 0.5f;
      xsn[t][2] = (x[p * 3 + 2] + 1.0f) * 0.5f;
    }
  }
  __syncthreads();
  if (t < 3) {                         // fill real W2 rows after zero-fill
    float s = 0.0f;
    for (int i = 0; i < 64; ++i) { float v = v2[t * 64 + i]; s += v * v; }
    float sc = g2[t] / sqrtf(s);
    for (int i = 0; i < 64; ++i) w2h[t][i] = (_Float16)(v2[t * 64 + i] * sc);
  }
  __syncthreads();

  // ---------- Load constant B matrices into registers (WMMA 16-bit B layout):
  // element e of lane -> B[k = (e&7) + (e>=8)*16 + hi*8][n = lane&15]; 2 chunks of 8 halfs.
  H16 B1[4], B2a, B2b;
  #pragma unroll
  for (int c = 0; c < 4; ++c) {
    const v8h* row = (const v8h*)&w1h[c * 16 + nIdx][0];   // 4 chunks of 8 halfs
    B1[c].h[0] = row[hi + 0];
    B1[c].h[1] = row[hi + 2];
  }
  {
    const v8h* row = (const v8h*)&w2h[nIdx][0];            // 8 chunks of 8 halfs
    B2a.h[0] = row[hi + 0];
    B2a.h[1] = row[hi + 2];
    B2b.h[0] = row[4 + hi + 0];
    B2b.h[1] = row[4 + hi + 2];
  }

  // ---------- Hash-grid encode: 16 points x 16 levels over 8 lane-iterations ----------
  #pragma unroll 1
  for (int it = 0; it < 8; ++it) {
    int w = it * 32 + lane;
    int p = w & 15;                 // point within this wave's tile
    int l = w >> 4;                 // level
    const float* xp = &xsn[wave * 16 + p][0];
    float rf = (float)P.res[l];
    float px = xp[0] * rf, py = xp[1] * rf, pz = xp[2] * rf;
    float fx = fminf(floorf(px), rf - 1.0f);
    float fy = fminf(floorf(py), rf - 1.0f);
    float fz = fminf(floorf(pz), rf - 1.0f);
    float frx = px - fx, fry = py - fy, frz = pz - fz;
    unsigned bx = (unsigned)fx, by = (unsigned)fy, bz = (unsigned)fz;
    const float* tb = tables + (size_t)l * (size_t)(TSIZE * 2);
    unsigned s1 = (unsigned)(P.res[l] + 1);
    bool dn = (P.dense[l] != 0);
    float f0 = 0.0f, f1 = 0.0f;
    #pragma unroll
    for (int c = 0; c < 8; ++c) {
      unsigned ox = (c >> 2) & 1, oy = (c >> 1) & 1, oz = c & 1;
      unsigned cx = bx + ox, cy = by + oy, cz = bz + oz;
      unsigned idx;
      if (dn) idx = cx + cy * s1 + cz * s1 * s1;
      else    idx = (cx ^ (cy * 2654435761u) ^ (cz * 805459861u)) & (TSIZE - 1u);
      float2 val = *(const float2*)(tb + (size_t)idx * 2);
      float wv = (ox ? frx : 1.0f - frx) * (oy ? fry : 1.0f - fry) * (oz ? frz : 1.0f - frz);
      f0 += wv * val.x;
      f1 += wv * val.y;
    }
    htile[wave][p][2 * l + 0] = (_Float16)f0;
    htile[wave][p][2 * l + 1] = (_Float16)f1;
  }
  __syncthreads();

  // ---------- Layer 1: [16,32] @ [32,64] via 4x v_wmma_f32_16x16x32_f16 ----------
  // 16-bit A layout: element e -> K = (e&7) + (e>=8)*16 + hi*8, M = lane&15.
  H16 A;
  {
    const v8h* row = (const v8h*)&htile[wave][nIdx][0];    // 4 chunks
    A.h[0] = row[hi + 0];
    A.h[1] = row[hi + 2];
  }
  #pragma unroll
  for (int c = 0; c < 4; ++c) {
    float bb = b1s[c * 16 + nIdx];     // bias is per-N: same for all 8 acc rows
    v8f acc;
    #pragma unroll
    for (int r = 0; r < 8; ++r) acc[r] = bb;
    acc = __builtin_amdgcn_wmma_f32_16x16x32_f16(false, A.v, false, B1[c].v,
                                                 (short)0, acc, false, false);
    #pragma unroll
    for (int r = 0; r < 8; ++r) {      // relu, write in [M][unit] layout
      float v = acc[r] > 0.0f ? acc[r] : 0.0f;
      r1t[wave][r + hi * 8][c * 16 + nIdx] = (_Float16)v;
    }
  }
  __syncthreads();

  // ---------- Layer 2: [16,64] @ [64,16 (3 used)] via 2x WMMA ----------
  H16 A2a, A2b;
  {
    const v8h* row = (const v8h*)&r1t[wave][nIdx][0];      // 8 chunks
    A2a.h[0] = row[hi + 0];
    A2a.h[1] = row[hi + 2];
    A2b.h[0] = row[hi + 4];
    A2b.h[1] = row[hi + 6];
  }
  float bb2 = b2s[nIdx < 3 ? nIdx : 3];
  v8f acc2;
  #pragma unroll
  for (int r = 0; r < 8; ++r) acc2[r] = bb2;
  acc2 = __builtin_amdgcn_wmma_f32_16x16x32_f16(false, A2a.v, false, B2a.v,
                                                (short)0, acc2, false, false);
  acc2 = __builtin_amdgcn_wmma_f32_16x16x32_f16(false, A2b.v, false, B2b.v,
                                                (short)0, acc2, false, false);

  const int tileBase = blockBase + wave * 16;
  if (nIdx < 3) {
    #pragma unroll
    for (int r = 0; r < 8; ++r) {
      int p = tileBase + r + hi * 8;
      if (p < npts) out[p * 3 + nIdx] = tanhf(acc2[r]);
    }
  }
}

extern "C" void kernel_launch(void* const* d_in, const int* in_sizes, int n_in,
                              void* d_out, int out_size, void* d_ws, size_t ws_size,
                              hipStream_t stream) {
  const float* x      = (const float*)d_in[0];
  const float* tables = (const float*)d_in[1];
  const float* v1 = (const float*)d_in[2];
  const float* g1 = (const float*)d_in[3];
  const float* b1 = (const float*)d_in[4];
  const float* v2 = (const float*)d_in[5];
  const float* g2 = (const float*)d_in[6];
  const float* b2 = (const float*)d_in[7];
  float* out = (float*)d_out;
  const int npts = in_sizes[0] / 3;

  // Replicate the reference's double-precision per-level resolution table.
  EncParams P;
  const double Bd = exp(log(512.0 / 16.0) / 15.0);
  for (int l = 0; l < NLVL; ++l) {
    int r = (int)floor(16.0 * pow(Bd, (double)l));
    P.res[l] = r;
    long long rp = (long long)r + 1;
    P.dense[l] = (rp * rp * rp <= (long long)TSIZE) ? 1 : 0;
  }

  const int nblocks = (npts + 127) / 128;   // N = 2^21 -> exactly 16384 full blocks
  ngp_fused_kernel<<<nblocks, 256, 0, stream>>>(x, tables, v1, g1, b1, v2, g2, b2,
                                                out, npts, P);
}
